// WaveletMixing2D_67568425500834
// MI455X (gfx1250) — matrix operand order repaired and checked by reading, not compiled, not run on hardware
//
#include <hip/hip_runtime.h>
#include <cstdint>

// ---------------------------------------------------------------------------
// Types
// ---------------------------------------------------------------------------
typedef unsigned int   u32;
typedef unsigned short u16;
typedef __attribute__((ext_vector_type(4)))  u32    u32x4;
typedef __attribute__((ext_vector_type(4)))  u32    v4u;
typedef __attribute__((ext_vector_type(8)))  int    v8i;
typedef __attribute__((ext_vector_type(4)))  int    v4i;
typedef __attribute__((ext_vector_type(16))) __bf16 v16bf;
typedef __attribute__((ext_vector_type(8)))  float  v8f;

#if __has_builtin(__builtin_amdgcn_tensor_load_to_lds)
#define HAVE_TDM 1
#else
#define HAVE_TDM 0
#endif

// db4 decomposition filters (pywt convention)
__constant__ float DEC_LO[8] = {
    -0.010597401784997278f,  0.032883011666982945f,  0.030841381835986965f,
    -0.18703481171888114f,  -0.02798376941698385f,   0.6308807679295904f,
     0.7148465705525415f,    0.23037781330885523f};
__constant__ float DEC_HI[8] = {
    -0.23037781330885523f,   0.7148465705525415f,   -0.6308807679295904f,
    -0.02798376941698385f,   0.18703481171888114f,   0.030841381835986965f,
    -0.032883011666982945f, -0.010597401784997278f};

__device__ __forceinline__ u16 f32_to_bf16(float f) {
    u32 u = __float_as_uint(f);
    u32 r = u + 0x7fffu + ((u >> 16) & 1u);   // round-to-nearest-even
    return (u16)(r >> 16);
}

// ---------------------------------------------------------------------------
// Forward DWT along last axis (W).  in: (rows, n) -> a,d: (rows, nc)
//   a[t] = sum_w DEC_LO[w] * x_sym(2t+1-w)   (symmetric extension)
// ---------------------------------------------------------------------------
__global__ void dwt_rows_kernel(const float* __restrict__ x,
                                float* __restrict__ a, float* __restrict__ d,
                                int rows, int n, int nc) {
    int idx = blockIdx.x * blockDim.x + threadIdx.x;
    if (idx >= rows * nc) return;
    int r = idx / nc, t = idx - r * nc;
    const float* xr = x + (size_t)r * n;
    float sa = 0.f, sd = 0.f;
#pragma unroll
    for (int w = 0; w < 8; ++w) {
        int i = 2 * t + 1 - w;
        i = (i < 0) ? (-1 - i) : i;
        i = (i >= n) ? (2 * n - 1 - i) : i;
        float v = xr[i];
        sa += v * DEC_LO[w];
        sd += v * DEC_HI[w];
    }
    a[idx] = sa;
    d[idx] = sd;
}

// Forward DWT along H.  in: (planes, n, Wd) -> a,d: (planes, nc, Wd)
__global__ void dwt_cols_kernel(const float* __restrict__ x,
                                float* __restrict__ a, float* __restrict__ d,
                                int planes, int n, int Wd, int nc) {
    int idx = blockIdx.x * blockDim.x + threadIdx.x;
    int total = planes * nc * Wd;
    if (idx >= total) return;
    int p = idx / (nc * Wd);
    int rem = idx - p * (nc * Wd);
    int t = rem / Wd, col = rem - t * Wd;
    const float* xp = x + (size_t)p * n * Wd + col;
    float sa = 0.f, sd = 0.f;
#pragma unroll
    for (int w = 0; w < 8; ++w) {
        int i = 2 * t + 1 - w;
        i = (i < 0) ? (-1 - i) : i;
        i = (i >= n) ? (2 * n - 1 - i) : i;
        float v = xp[(size_t)i * Wd];
        sa += v * DEC_LO[w];
        sd += v * DEC_HI[w];
    }
    a[idx] = sa;
    d[idx] = sd;
}

// ---------------------------------------------------------------------------
// Inverse DWT (zero-padded upsample + filter, crop to nout):
//   y[i] = sum_j a[j]*DEC_LO[2j-i+1] + d[j]*DEC_HI[2j-i+1], tap in [0,7]
// ---------------------------------------------------------------------------
__global__ void idwt_cols_kernel(const float* __restrict__ a, const float* __restrict__ d,
                                 float* __restrict__ y,
                                 int planes, int nc, int Wd, int nout) {
    int idx = blockIdx.x * blockDim.x + threadIdx.x;
    int total = planes * nout * Wd;
    if (idx >= total) return;
    int p = idx / (nout * Wd);
    int rem = idx - p * (nout * Wd);
    int i = rem / Wd, col = rem - i * Wd;
    const float* ap = a + (size_t)p * nc * Wd + col;
    const float* dp = d + (size_t)p * nc * Wd + col;
    int j0 = i >> 1;
    int j1 = (i + 6) >> 1; if (j1 > nc - 1) j1 = nc - 1;
    float s = 0.f;
    for (int j = j0; j <= j1; ++j) {
        int w = 2 * j - i + 1;
        s += ap[(size_t)j * Wd] * DEC_LO[w] + dp[(size_t)j * Wd] * DEC_HI[w];
    }
    y[idx] = s;
}

__global__ void idwt_rows_kernel(const float* __restrict__ a, const float* __restrict__ d,
                                 float* __restrict__ y, int rows, int nc, int nout) {
    int idx = blockIdx.x * blockDim.x + threadIdx.x;
    if (idx >= rows * nout) return;
    int r = idx / nout, i = idx - r * nout;
    const float* ar = a + (size_t)r * nc;
    const float* dr = d + (size_t)r * nc;
    int j0 = i >> 1;
    int j1 = (i + 6) >> 1; if (j1 > nc - 1) j1 = nc - 1;
    float s = 0.f;
    for (int j = j0; j <= j1; ++j) {
        int w = 2 * j - i + 1;
        s += ar[j] * DEC_LO[w] + dr[j] * DEC_HI[w];
    }
    y[idx] = s;
}

// Final level-1 row IDWT fused with residual add; writes d_out.
__global__ void idwt_rows_add_kernel(const float* __restrict__ a, const float* __restrict__ d,
                                     const float* __restrict__ x, float* __restrict__ out,
                                     int rows, int nc, int nout) {
    int idx = blockIdx.x * blockDim.x + threadIdx.x;
    if (idx >= rows * nout) return;
    int r = idx / nout, i = idx - r * nout;
    const float* ar = a + (size_t)r * nc;
    const float* dr = d + (size_t)r * nc;
    int j0 = i >> 1;
    int j1 = (i + 6) >> 1; if (j1 > nc - 1) j1 = nc - 1;
    float s = 0.f;
    for (int j = j0; j <= j1; ++j) {
        int w = 2 * j - i + 1;
        s += ar[j] * DEC_LO[w] + dr[j] * DEC_HI[w];
    }
    out[idx] = s + x[idx];
}

// ---------------------------------------------------------------------------
// Packing: f32 NCHW -> bf16 NHWC (channels contiguous for implicit GEMM K)
// ---------------------------------------------------------------------------
__global__ void pack_nhwc_bf16_kernel(const float* __restrict__ in, u16* __restrict__ out,
                                      int HW) {
    int idx = blockIdx.x * blockDim.x + threadIdx.x;
    int total = 32 * 256 * HW;
    if (idx >= total) return;
    int b = idx / (256 * HW);
    int rem = idx - b * (256 * HW);
    int c = rem / HW;
    int s = rem - c * HW;
    out[((size_t)b * HW + s) * 256 + c] = f32_to_bf16(in[idx]);
}

// W[oc][ic][kh][kw] f32 -> Wp[oc][(kh*3+kw)*256 + ic] bf16
__global__ void pack_weights_kernel(const float* __restrict__ w, u16* __restrict__ out) {
    int idx = blockIdx.x * blockDim.x + threadIdx.x;
    if (idx >= 256 * 2304) return;
    int oc = idx / 2304;
    int k = idx - oc * 2304;
    int p = k >> 8;          // kh*3+kw
    int ic = k & 255;
    out[idx] = f32_to_bf16(w[((size_t)(oc * 256 + ic)) * 9 + p]);
}

// ---------------------------------------------------------------------------
// Implicit-GEMM 3x3 conv, C=256 -> 256, pad 1, via v_wmma_f32_16x16x32_bf16.
//   M = 32*H*W spatial, N = 256 out-channels, K = 9*256 = 2304.
//   Block: 256 threads (8 wave32), 128x64 output tile, 32x32 per wave
//   (4 accumulators), LDS double-buffered.
//   A tile 128x32 bf16 gathered from NHWC activations (pad -> zeros) with
//   plain VMEM (needs zero-fill).  B tile 64(n)x32(k) bf16 is a dense
//   constant-stride 2-D tile: fed by the Tensor Data Mover
//   (tensor_load_to_lds, wave-0 issued, TENSORcnt-fenced, LDS row padding
//   reproduced via D# pad_interval/pad_amount).
// ---------------------------------------------------------------------------
#define CONV_K 2304

__global__ __launch_bounds__(256) void conv3x3_wmma_kernel(
    const u16* __restrict__ xin,   // bf16 NHWC  [32][Hh][Ww][256]
    const u16* __restrict__ wp,    // bf16       [256][2304]
    const float* __restrict__ bias,
    float* __restrict__ out,       // f32 NCHW   [32][256][Hh][Ww]
    int Hh, int Ww) {

    const int HW  = Hh * Ww;
    const int Msz = 32 * HW;
    const int m0  = blockIdx.x * 128;
    const int n0  = blockIdx.y * 64;

    __shared__ u16 sA[2][128][40];   // row stride padded to 80B
    __shared__ u16 sB[2][64][40];    // [n][k] layout, same padding

    const int tid  = threadIdx.x;
    const int wid  = tid >> 5;
    const int lane = tid & 31;
    const int mw   = wid >> 1;      // wave M-tile: 32*mw
    const int nw   = wid & 1;       // wave N-tile: 32*nw

    // loader mapping: thread -> (rows lrow, lrow+64; 16B chunk 0..3)
    const int lrow = tid >> 2;
    const int lq   = tid & 3;

    // decode the two A-row pixels once
    int mA[2], bA[2], hA[2], wA[2];
#pragma unroll
    for (int i = 0; i < 2; ++i) {
        int m = m0 + lrow + i * 64;
        mA[i] = m;
        int b = m / HW;
        int rm = m - b * HW;
        int hh = rm / Ww;
        bA[i] = b; hA[i] = hh; wA[i] = rm - hh * Ww;
    }

    v8f acc00 = {}, acc01 = {}, acc10 = {}, acc11 = {};

    auto loadA = [&](int kc, int buf) {
        const int p   = kc >> 8;            // tap (kh*3+kw), fixed per 32-chunk
        const int kh  = p / 3;
        const int kw  = p - kh * 3;
        const int ic0 = (kc & 255) + lq * 8;
#pragma unroll
        for (int i = 0; i < 2; ++i) {
            const int ah = hA[i] + kh - 1;
            const int aw = wA[i] + kw - 1;
            u32x4 v = {};
            if (mA[i] < Msz && (unsigned)ah < (unsigned)Hh && (unsigned)aw < (unsigned)Ww) {
                v = *reinterpret_cast<const u32x4*>(
                    xin + (((size_t)bA[i] * Hh + ah) * Ww + aw) * 256 + ic0);
            }
            *reinterpret_cast<u32x4*>(&sA[buf][lrow + i * 64][lq * 8]) = v;
        }
    };

#if HAVE_TDM
    // Tensor Data Mover: DMA the 64(n) x 32(k) bf16 weight tile into sB[buf].
    // D# group0: count=1, lds_addr, global_addr, type=2.  group1: data_size=2B,
    // tensor 2304x256 stride 2304, tile 32x64, pad 4 DWORDs every 16 DWORDs
    // (reproduces the 40-element padded LDS rows).
    auto tdmB = [&](int kc, int buf) {
        unsigned long long ga =
            (unsigned long long)(size_t)(wp + (size_t)n0 * CONV_K + kc);
        u32 lds_off = (u32)(uintptr_t)(&sB[buf][0][0]);
        v4u g0;
        g0[0] = 1u;                                  // count=1 (user descriptor)
        g0[1] = lds_off;                             // lds_addr (bytes)
        g0[2] = (u32)ga;                             // global_addr[31:0]
        g0[3] = (u32)(ga >> 32) | 0x80000000u;       // global_addr[56:32] | type=2
        v8i g1;
        g1[0] = (int)((1u << 16) |                   // data_size = 2 bytes
                      (1u << 20) |                   // pad_enable
                      (3u << 22) |                   // pad_interval: 16 DWORDs
                      (3u << 25));                   // pad_amount:   4 DWORDs
        g1[1] = (int)((CONV_K & 0xffffu) << 16);     // tensor_dim0[15:0]
        g1[2] = (int)((CONV_K >> 16) | (256u << 16));// tensor_dim0[31:16] | dim1[15:0]
        g1[3] = (int)(32u << 16);                    // dim1[31:16]=0 | tile_dim0=32
        g1[4] = (int)64u;                            // tile_dim1=64, tile_dim2=0
        g1[5] = (int)CONV_K;                         // tensor_dim0_stride[31:0]
        g1[6] = 0;                                   // stride hi | dim1_stride lo
        g1[7] = 0;
        v4i gz = {0, 0, 0, 0};
#if defined(__clang_major__) && (__clang_major__ >= 23)
        v8i gz8 = {0, 0, 0, 0, 0, 0, 0, 0};
        __builtin_amdgcn_tensor_load_to_lds(g0, g1, gz, gz, gz8, 0);
#else
        __builtin_amdgcn_tensor_load_to_lds(g0, g1, gz, gz, 0);
#endif
    };
#else
    auto loadB = [&](int kc, int buf) {
        const int n = n0 + (tid >> 2);
        u32x4 vb = *reinterpret_cast<const u32x4*>(
            wp + (size_t)n * CONV_K + kc + lq * 8);
        *reinterpret_cast<u32x4*>(&sB[buf][tid >> 2][lq * 8]) = vb;
    };
#endif

    // fragment addressing per ISA 7.12.2 (wave32, 16-bit 16x32 A / 32x16 B)
    const int fm  = lane & 15;
    const int kh2 = lane >> 4;

    auto wmma_step = [&](int buf) {
        union Frag { u32x4 u[2]; v16bf v; };
        Frag a0, a1, b0, b1;
        a0.u[0] = *reinterpret_cast<const u32x4*>(&sA[buf][mw * 32 + fm][kh2 * 8]);
        a0.u[1] = *reinterpret_cast<const u32x4*>(&sA[buf][mw * 32 + fm][16 + kh2 * 8]);
        a1.u[0] = *reinterpret_cast<const u32x4*>(&sA[buf][mw * 32 + 16 + fm][kh2 * 8]);
        a1.u[1] = *reinterpret_cast<const u32x4*>(&sA[buf][mw * 32 + 16 + fm][16 + kh2 * 8]);
        b0.u[0] = *reinterpret_cast<const u32x4*>(&sB[buf][nw * 32 + fm][kh2 * 8]);
        b0.u[1] = *reinterpret_cast<const u32x4*>(&sB[buf][nw * 32 + fm][16 + kh2 * 8]);
        b1.u[0] = *reinterpret_cast<const u32x4*>(&sB[buf][nw * 32 + 16 + fm][kh2 * 8]);
        b1.u[1] = *reinterpret_cast<const u32x4*>(&sB[buf][nw * 32 + 16 + fm][16 + kh2 * 8]);
        acc00 = __builtin_amdgcn_wmma_f32_16x16x32_bf16(
            false, a0.v, false, b0.v, (short)0, acc00, false, false);
        acc01 = __builtin_amdgcn_wmma_f32_16x16x32_bf16(
            false, a0.v, false, b1.v, (short)0, acc01, false, false);
        acc10 = __builtin_amdgcn_wmma_f32_16x16x32_bf16(
            false, a1.v, false, b0.v, (short)0, acc10, false, false);
        acc11 = __builtin_amdgcn_wmma_f32_16x16x32_bf16(
            false, a1.v, false, b1.v, (short)0, acc11, false, false);
    };

    // prologue: fill buffer 0
#if HAVE_TDM
    if (wid == 0) tdmB(0, 0);
#else
    loadB(0, 0);
#endif
    loadA(0, 0);
#if HAVE_TDM
    if (wid == 0) __builtin_amdgcn_s_wait_tensorcnt(0);
#endif
    __syncthreads();

#pragma unroll 1
    for (int kc = 0; kc < CONV_K; kc += 32) {
        const int buf = (kc >> 5) & 1;
        if (kc + 32 < CONV_K) {
#if HAVE_TDM
            if (wid == 0) tdmB(kc + 32, buf ^ 1);   // DMA overlaps the WMMA stage
#else
            loadB(kc + 32, buf ^ 1);
#endif
            loadA(kc + 32, buf ^ 1);
        }
        wmma_step(buf);
#if HAVE_TDM
        if (wid == 0) __builtin_amdgcn_s_wait_tensorcnt(0);
#endif
        __syncthreads();
    }

    // store (C/D layout: lanes 0-15 -> M=r, lanes 16-31 -> M=r+8; N = lane&15)
    const v8f* accs[2][2] = {{&acc00, &acc01}, {&acc10, &acc11}};
#pragma unroll
    for (int mi = 0; mi < 2; ++mi) {
#pragma unroll
        for (int ni = 0; ni < 2; ++ni) {
            const int nn = n0 + nw * 32 + ni * 16 + fm;
            const float bs = bias[nn];
            const v8f& a = *accs[mi][ni];
#pragma unroll
            for (int r = 0; r < 8; ++r) {
                const int mg = m0 + mw * 32 + mi * 16 + kh2 * 8 + r;
                if (mg < Msz) {
                    const int bb = mg / HW;
                    const int rr = mg - bb * HW;
                    const int hy = rr / Ww;
                    const int wy = rr - hy * Ww;
                    out[(((size_t)bb * 256 + nn) * Hh + hy) * Ww + wy] = a[r] + bs;
                }
            }
        }
    }
}

// ---------------------------------------------------------------------------
// BatchNorm statistics over (B,H,W) per channel (biased variance) + apply/ReLU
// ---------------------------------------------------------------------------
__global__ void bn_stats_kernel(const float* __restrict__ y,
                                float* __restrict__ mean, float* __restrict__ rstd,
                                int HW) {
    const int c = blockIdx.x;   // 0..255
    const int N = 32 * HW;
    float s = 0.f, q = 0.f;
    for (int i = threadIdx.x; i < N; i += 256) {
        int b = i / HW, sp = i - b * HW;
        float v = y[((size_t)b * 256 + c) * HW + sp];
        s += v;
        q += v * v;
    }
    __shared__ float ss[256], sq[256];
    ss[threadIdx.x] = s;
    sq[threadIdx.x] = q;
    __syncthreads();
    for (int st = 128; st > 0; st >>= 1) {
        if (threadIdx.x < st) {
            ss[threadIdx.x] += ss[threadIdx.x + st];
            sq[threadIdx.x] += sq[threadIdx.x + st];
        }
        __syncthreads();
    }
    if (threadIdx.x == 0) {
        float mu  = ss[0] / (float)N;
        float var = sq[0] / (float)N - mu * mu;
        mean[c] = mu;
        rstd[c] = rsqrtf(var + 1e-5f);
    }
}

__global__ void bn_apply_relu_kernel(float* __restrict__ y,
                                     const float* __restrict__ mean,
                                     const float* __restrict__ rstd,
                                     const float* __restrict__ g,
                                     const float* __restrict__ bta,
                                     int HW) {
    int idx = blockIdx.x * blockDim.x + threadIdx.x;
    int total = 32 * 256 * HW;
    if (idx >= total) return;
    int c = (idx / HW) & 255;
    float v = (y[idx] - mean[c]) * rstd[c] * g[c] + bta[c];
    y[idx] = v > 0.f ? v : 0.f;
}

// ---------------------------------------------------------------------------
// Host-side orchestration
// ---------------------------------------------------------------------------
static inline int cdiv_i(long a, long b) { return (int)((a + b - 1) / b); }

extern "C" void kernel_launch(void* const* d_in, const int* in_sizes, int n_in,
                              void* d_out, int out_size, void* d_ws, size_t ws_size,
                              hipStream_t stream) {
    (void)in_sizes; (void)n_in; (void)out_size; (void)ws_size;

    const float* x     = (const float*)d_in[0];
    const float* ll_w  = (const float*)d_in[1];
    const float* ll_b  = (const float*)d_in[2];
    const float* bn_g  = (const float*)d_in[3];
    const float* bn_b  = (const float*)d_in[4];
    const float* det_w = (const float*)d_in[5];
    const float* det_b = (const float*)d_in[6];
    float* outp = (float*)d_out;

    constexpr int Bb = 32, Cc = 256;
    constexpr int H0 = 64, W0 = 64;
    constexpr int H1 = 35, W1 = 35;   // level-1 coeff size
    constexpr int H2 = 21, W2 = 21;   // level-2 coeff size
    constexpr int PL = Bb * Cc;       // planes

    const size_t N_A1 = (size_t)PL * H0 * W1;  // (64,35)
    const size_t N_L1 = (size_t)PL * H1 * W1;  // (35,35)
    const size_t N_A2 = (size_t)PL * H1 * W2;  // (35,21)
    const size_t N_L2 = (size_t)PL * H2 * W2;  // (21,21)

    char* wsp = (char*)d_ws;
    auto alloc = [&](size_t bytes) -> char* {
        char* r = wsp;
        wsp += (bytes + 255) & ~(size_t)255;
        return r;
    };

    // f32 scratch (time-aliased buffers share allocations)
    float* a1   = (float*)alloc(N_A1 * 4);   // also reused as lo1r
    float* d1   = (float*)alloc(N_A1 * 4);   // also reused as hi1r
    float* ll1  = (float*)alloc(N_L1 * 4);   // also reused as llr1
    float* lh1  = (float*)alloc(N_L1 * 4);   // also reused as ylh1
    float* hl1  = (float*)alloc(N_L1 * 4);   // also reused as yhl1
    float* hh1  = (float*)alloc(N_L1 * 4);   // also reused as yhh1
    float* a2   = (float*)alloc(N_A2 * 4);   // also reused as lo2r
    float* d2   = (float*)alloc(N_A2 * 4);   // also reused as hi2r
    float* ll2  = (float*)alloc(N_L2 * 4);   // also reused as yll2
    float* lh2  = (float*)alloc(N_L2 * 4);   // also reused as ylh2
    float* hl2  = (float*)alloc(N_L2 * 4);   // also reused as yhl2
    float* hh2  = (float*)alloc(N_L2 * 4);   // also reused as yhh2
    float* bnm  = (float*)alloc(256 * 4);
    float* bnr  = (float*)alloc(256 * 4);

    // bf16 buffers
    u16* ll2b = (u16*)alloc(N_L2 * 2);
    u16* lh2b = (u16*)alloc(N_L2 * 2);
    u16* hl2b = (u16*)alloc(N_L2 * 2);
    u16* hh2b = (u16*)alloc(N_L2 * 2);
    u16* lh1b = (u16*)alloc(N_L1 * 2);
    u16* hl1b = (u16*)alloc(N_L1 * 2);
    u16* hh1b = (u16*)alloc(N_L1 * 2);
    u16* wpk  = (u16*)alloc((size_t)7 * 256 * 2304 * 2);   // [7][256][2304]

    const dim3 blk(256);
    auto g1 = [&](long total) { return dim3((unsigned)cdiv_i(total, 256)); };

    // ---- Forward DWT, level 1 ----
    dwt_rows_kernel<<<g1((long)PL * H0 * W1), blk, 0, stream>>>(x, a1, d1, PL * H0, W0, W1);
    dwt_cols_kernel<<<g1((long)PL * H1 * W1), blk, 0, stream>>>(a1, ll1, lh1, PL, H0, W1, H1);
    dwt_cols_kernel<<<g1((long)PL * H1 * W1), blk, 0, stream>>>(d1, hl1, hh1, PL, H0, W1, H1);
    // ---- Forward DWT, level 2 (on ll1) ----
    dwt_rows_kernel<<<g1((long)PL * H1 * W2), blk, 0, stream>>>(ll1, a2, d2, PL * H1, W1, W2);
    dwt_cols_kernel<<<g1((long)PL * H2 * W2), blk, 0, stream>>>(a2, ll2, lh2, PL, H1, W2, H2);
    dwt_cols_kernel<<<g1((long)PL * H2 * W2), blk, 0, stream>>>(d2, hl2, hh2, PL, H1, W2, H2);

    // ---- Pack activations to bf16 NHWC ----
    const int HW1 = H1 * W1, HW2 = H2 * W2;
    pack_nhwc_bf16_kernel<<<g1((long)N_L2), blk, 0, stream>>>(ll2, ll2b, HW2);
    pack_nhwc_bf16_kernel<<<g1((long)N_L2), blk, 0, stream>>>(lh2, lh2b, HW2);
    pack_nhwc_bf16_kernel<<<g1((long)N_L2), blk, 0, stream>>>(hl2, hl2b, HW2);
    pack_nhwc_bf16_kernel<<<g1((long)N_L2), blk, 0, stream>>>(hh2, hh2b, HW2);
    pack_nhwc_bf16_kernel<<<g1((long)N_L1), blk, 0, stream>>>(lh1, lh1b, HW1);
    pack_nhwc_bf16_kernel<<<g1((long)N_L1), blk, 0, stream>>>(hl1, hl1b, HW1);
    pack_nhwc_bf16_kernel<<<g1((long)N_L1), blk, 0, stream>>>(hh1, hh1b, HW1);

    // ---- Pack weights to bf16 [oc][k]; slot 0 = ll, 1+l*3+i = det_w[l][i] ----
    const size_t WSZ = (size_t)256 * 2304;
    const size_t WEL = (size_t)256 * 256 * 9;
    pack_weights_kernel<<<g1((long)WSZ), blk, 0, stream>>>(ll_w, wpk);
    for (int l = 0; l < 2; ++l)
        for (int i = 0; i < 3; ++i)
            pack_weights_kernel<<<g1((long)WSZ), blk, 0, stream>>>(
                det_w + (size_t)(l * 3 + i) * WEL, wpk + (size_t)(1 + l * 3 + i) * WSZ);

    // ---- WMMA convs ----
    auto conv = [&](const u16* in, int slot, const float* bias, float* out_,
                    int Hh, int Ww) {
        const int M = 32 * Hh * Ww;
        dim3 grid((unsigned)cdiv_i(M, 128), 4);
        conv3x3_wmma_kernel<<<grid, blk, 0, stream>>>(in, wpk + (size_t)slot * WSZ,
                                                      bias, out_, Hh, Ww);
    };
    conv(ll2b, 0, ll_b, ll2, H2, W2);                      // yll2 (in-place alias of ll2)
    // BN + ReLU on LL branch
    bn_stats_kernel<<<dim3(256), blk, 0, stream>>>(ll2, bnm, bnr, HW2);
    bn_apply_relu_kernel<<<g1((long)N_L2), blk, 0, stream>>>(ll2, bnm, bnr, bn_g, bn_b, HW2);
    // detail convs: details[1] = level-2 bands, details[0] = level-1 bands
    conv(lh2b, 1 + 1 * 3 + 0, det_b + (1 * 3 + 0) * 256, lh2, H2, W2);
    conv(hl2b, 1 + 1 * 3 + 1, det_b + (1 * 3 + 1) * 256, hl2, H2, W2);
    conv(hh2b, 1 + 1 * 3 + 2, det_b + (1 * 3 + 2) * 256, hh2, H2, W2);
    conv(lh1b, 1 + 0 * 3 + 0, det_b + (0 * 3 + 0) * 256, lh1, H1, W1);
    conv(hl1b, 1 + 0 * 3 + 1, det_b + (0 * 3 + 1) * 256, hl1, H1, W1);
    conv(hh1b, 1 + 0 * 3 + 2, det_b + (0 * 3 + 2) * 256, hh1, H1, W1);

    // ---- Inverse DWT, level 2 ----
    idwt_cols_kernel<<<g1((long)N_A2), blk, 0, stream>>>(ll2, lh2, a2, PL, H2, W2, H1); // lo2r
    idwt_cols_kernel<<<g1((long)N_A2), blk, 0, stream>>>(hl2, hh2, d2, PL, H2, W2, H1); // hi2r
    idwt_rows_kernel<<<g1((long)N_L1), blk, 0, stream>>>(a2, d2, ll1, PL * H1, W2, W1); // llr1
    // ---- Inverse DWT, level 1 ----
    idwt_cols_kernel<<<g1((long)N_A1), blk, 0, stream>>>(ll1, lh1, a1, PL, H1, W1, H0); // lo1r
    idwt_cols_kernel<<<g1((long)N_A1), blk, 0, stream>>>(hl1, hh1, d1, PL, H1, W1, H0); // hi1r
    // final row IDWT fused with residual add -> d_out
    idwt_rows_add_kernel<<<g1((long)PL * H0 * W0), blk, 0, stream>>>(
        a1, d1, x, outp, PL * H0, W1, W0);
}
